// MRNN_10840497455548
// MI455X (gfx1250) — compile-verified, matrix operand order checked by reading, not run on hardware
//
#include <hip/hip_runtime.h>
#include <hip/hip_bf16.h>

// ---------------------------------------------------------------------------
// MRNN (bidirectional GRU + FCN regression) for MI455X / gfx1250.
// bf16 WMMA (v_wmma_f32_16x16x32_bf16) everywhere, f32 accumulation,
// f32 recurrent state. All GEMMs use a 128x64 block tile, BK=32, 8 waves.
// LDS staging uses GLOBAL_LOAD_ASYNC_TO_LDS_B128 (ASYNCcnt) when the
// toolchain exposes the builtins, else global_load_b128 + ds_store_b128.
// ---------------------------------------------------------------------------

typedef __bf16 bf16;
typedef __attribute__((ext_vector_type(16))) __bf16 v16bf;
typedef __attribute__((ext_vector_type(8)))  float  v8f;

#define Bsz 512
#define Tsz 64
#define Fsz 256
#define Hsz 512
#define MROWS (Tsz * Bsz)     // 32768 time-major rows

#define BM 128
#define BN 64
#define BK 32
#define LDSP (BK + 8)         // 40 bf16 -> 80B row pitch (16B aligned)
#define NTH 256

#if __has_builtin(__builtin_amdgcn_global_load_async_to_lds_b128) && \
    __has_builtin(__builtin_amdgcn_s_wait_asynccnt)
#define USE_ASYNC_LDS 1
#else
#define USE_ASYNC_LDS 0
#endif

#if USE_ASYNC_LDS
// Builtin signature (from hipcc diagnostic): param0 is
// '__attribute__((vector_size(16))) int __device__ *'  (AS1 = global).
typedef int v4i_vs __attribute__((vector_size(16)));
typedef __attribute__((address_space(1))) v4i_vs gv4i;
typedef __attribute__((address_space(3))) v4i_vs lv4i;
// one GLOBAL_LOAD_ASYNC_TO_LDS_B128: 16 bytes, global -> LDS, no VGPR data.
// Generic->AS casts go through integers (inttoptr): AS1 keeps the 64-bit
// value; AS3 is the low-32-bit LDS offset of the generic pointer.
__device__ __forceinline__ void async_copy16(const bf16* g, bf16* l) {
  __builtin_amdgcn_global_load_async_to_lds_b128(
      (gv4i*)(unsigned long long)g,
      (lv4i*)(unsigned int)(unsigned long long)l, 0, 0);
}
__device__ __forceinline__ void wait_async0() {
  __builtin_amdgcn_s_wait_asynccnt(0);
}
#endif

__device__ __forceinline__ v8f vzero() {
  v8f v;
#pragma unroll
  for (int i = 0; i < 8; ++i) v[i] = 0.f;
  return v;
}

__device__ __forceinline__ void wmma_acc(v8f& c, const v16bf& a, const v16bf& b) {
  // D = A*B + C, bf16 inputs, f32 accumulate (CDNA5 wave32 WMMA)
  c = __builtin_amdgcn_wmma_f32_16x16x32_bf16(false, a, false, b, (short)0, c,
                                              false, false);
}

// A-frag (16x32) / B-frag (32x16) per ISA layout: lanes<16 hold K 0..7,16..23;
// lanes>=16 hold K 8..15,24..31. Two 16B LDS loads per fragment.
__device__ __forceinline__ v16bf frag_from_lds(const bf16* row, int khalf) {
  union { uint4 u[2]; v16bf f; } t;
  t.u[0] = *reinterpret_cast<const uint4*>(row + khalf);
  t.u[1] = *reinterpret_cast<const uint4*>(row + 16 + khalf);
  return t.f;
}

// C layout: VGPR e, lanes0-15 -> M=e, lanes16-31 -> M=8+e; N = lane&15
__device__ __forceinline__ int acc_row(int tid, int i, int e) {
  int wave = tid >> 5, lane = tid & 31;
  return ((wave & 3) << 5) + (i << 4) + ((lane >> 4) << 3) + e;
}
__device__ __forceinline__ int acc_col(int tid, int j) {
  int wave = tid >> 5, lane = tid & 31;
  return ((wave >> 2) << 5) + (j << 4) + (lane & 15);
}

// stage one 128x32 A tile + one 64x32 B tile into LDS (per K-step)
__device__ __forceinline__ void stage_tiles(const bf16* __restrict__ At, long lda,
                                            const bf16* __restrict__ Wt, long ldw,
                                            int k0, bf16* As, bf16* Bs) {
  const int tid = threadIdx.x;
#if USE_ASYNC_LDS
  {  // A tile: 32B per thread = 2 async b128 copies
    int r = tid >> 1, c = (tid & 1) << 4;
    const bf16* g = At + (long)r * lda + k0 + c;
    bf16* l = As + r * LDSP + c;
    async_copy16(g, l);
    async_copy16(g + 8, l + 8);
  }
  {  // B tile: 16B per thread
    int r = tid >> 2, c = (tid & 3) << 3;
    async_copy16(Wt + (long)r * ldw + k0 + c, Bs + r * LDSP + c);
  }
#else
  {
    int r = tid >> 1, c = (tid & 1) << 4;
    const uint4* s = reinterpret_cast<const uint4*>(At + (long)r * lda + k0 + c);
    uint4* d = reinterpret_cast<uint4*>(As + r * LDSP + c);
    d[0] = s[0]; d[1] = s[1];
  }
  {
    int r = tid >> 2, c = (tid & 3) << 3;
    *reinterpret_cast<uint4*>(Bs + r * LDSP + c) =
        *reinterpret_cast<const uint4*>(Wt + (long)r * ldw + k0 + c);
  }
#endif
}

__device__ __forceinline__ void stage_fence() {
#if USE_ASYNC_LDS
  wait_async0();   // each wave waits for its own async copies...
#endif
  __syncthreads(); // ...then the workgroup barrier publishes the tiles
}

// acc[2][2] += A[0:128, 0:K] * W[0:64, 0:K]^T   (A, W bf16 row-major)
__device__ __forceinline__ void gemm_core(const bf16* __restrict__ At, long lda,
                                          const bf16* __restrict__ Wt, long ldw,
                                          int K, v8f acc[2][2],
                                          bf16* As, bf16* Bs) {
  const int tid  = threadIdx.x;
  const int lane = tid & 31;
  const int wave = tid >> 5;
  const int arow  = ((wave & 3) << 5) + (lane & 15);
  const int bcol  = ((wave >> 2) << 5) + (lane & 15);
  const int khalf = (lane >> 4) << 3;
  for (int k0 = 0; k0 < K; k0 += BK) {
    __syncthreads();
    stage_tiles(At, lda, Wt, ldw, k0, As, Bs);
    if (k0 + BK < K)  // global_prefetch next A tile
      __builtin_prefetch(At + (long)(tid >> 1) * lda + k0 + BK, 0, 1);
    stage_fence();
    v16bf a0 = frag_from_lds(As + arow * LDSP, khalf);
    v16bf a1 = frag_from_lds(As + (arow + 16) * LDSP, khalf);
    v16bf b0 = frag_from_lds(Bs + bcol * LDSP, khalf);
    v16bf b1 = frag_from_lds(Bs + (bcol + 16) * LDSP, khalf);
    wmma_acc(acc[0][0], a0, b0);
    wmma_acc(acc[0][1], a0, b1);
    wmma_acc(acc[1][0], a1, b0);
    wmma_acc(acc[1][1], a1, b1);
  }
  __syncthreads();
}

__device__ __forceinline__ float sigmoidf_(float x) {
  return 1.f / (1.f + __expf(-x));
}

// ------------------------- data preparation --------------------------------

// Build time-major bf16 GRU inputs xs[dir][T*B, 3F] = [X | mask | delta],
// and bf16 copies of x, m (forward) for the FCN GEMMs.
__global__ void k_pack_xs(const float* __restrict__ Xf, const float* __restrict__ Mf,
                          const float* __restrict__ Df, const float* __restrict__ Xb,
                          const float* __restrict__ Mb, const float* __restrict__ Db,
                          bf16* __restrict__ xs, bf16* __restrict__ x_bf,
                          bf16* __restrict__ m_bf) {
  int dir = blockIdx.z;
  long idx = (long)blockIdx.x * blockDim.x + threadIdx.x;
  if (idx >= (long)MROWS * 3 * Fsz) return;
  int col = (int)(idx % (3 * Fsz));
  long row = idx / (3 * Fsz);
  int b = (int)(row % Bsz), t = (int)(row / Bsz);
  int seg = col >> 8, f = col & 255;
  const float* src =
      dir ? (seg == 0 ? Xb : (seg == 1 ? Mb : Db))
          : (seg == 0 ? Xf : (seg == 1 ? Mf : Df));
  float v = src[((long)b * Tsz + t) * Fsz + f];
  xs[(long)dir * MROWS * 3 * Fsz + idx] = (bf16)v;
  if (dir == 0) {
    if (seg == 0) x_bf[row * Fsz + f] = (bf16)v;
    else if (seg == 1) m_bf[row * Fsz + f] = (bf16)v;
  }
}

__global__ void k_cvt(const float* __restrict__ src, bf16* __restrict__ dst,
                      int rows, int cols, int zero_diag) {
  long i = (long)blockIdx.x * blockDim.x + threadIdx.x;
  if (i >= (long)rows * cols) return;
  int r = (int)(i / cols), c = (int)(i % cols);
  float v = src[i];
  if (zero_diag && r == c) v = 0.f;  // (U*mm), (V1*mm): off-diagonal mask
  dst[i] = (bf16)v;
}

__global__ void k_zero(float* hf, bf16* hb, float* sums, int nsum) {
  long i = (long)blockIdx.x * blockDim.x + threadIdx.x;
  const long nh = 2L * 2 * Bsz * Hsz;
  if (i < nh) { hf[i] = 0.f; hb[i] = (bf16)0.f; }
  if (i < nsum) sums[i] = 0.f;
}

// ------------------------- GEMM kernels ------------------------------------

// gi[dir] = xs[dir] @ Wih[dir]^T + bih[dir]   : [T*B, 3H] f32, K=3F=768
__global__ void k_gi(const bf16* __restrict__ xs, const bf16* __restrict__ Wih_bf,
                     const float* __restrict__ bih_f, const float* __restrict__ bih_b,
                     float* __restrict__ gi) {
  __shared__ bf16 As[BM * LDSP];
  __shared__ bf16 Bs[BN * LDSP];
  int dir = blockIdx.z;
  long m0 = (long)blockIdx.x * BM;
  int n0 = blockIdx.y * BN;
  v8f acc[2][2];
#pragma unroll
  for (int i = 0; i < 2; ++i)
#pragma unroll
    for (int j = 0; j < 2; ++j) acc[i][j] = vzero();
  gemm_core(xs + (long)dir * MROWS * 768 + m0 * 768, 768,
            Wih_bf + (long)dir * 1536 * 768 + (long)n0 * 768, 768, 768,
            acc, As, Bs);
  const float* bias = dir ? bih_b : bih_f;
  float* out = gi + (long)dir * MROWS * 1536;
  int tid = threadIdx.x;
#pragma unroll
  for (int i = 0; i < 2; ++i)
#pragma unroll
    for (int j = 0; j < 2; ++j) {
      int c = n0 + acc_col(tid, j);
      float bv = bias[c];
#pragma unroll
      for (int e = 0; e < 8; ++e) {
        long r = m0 + acc_row(tid, i, e);
        out[r * 1536 + c] = acc[i][j][e] + bv;
      }
    }
}

// One fused GRU timestep (both directions via blockIdx.z):
//   gh_g = h_prev @ Whh_g^T (3 gates, shared A tile), gate math, h update,
//   scatter bf16 h into hcat[T*B, 2H] (backward dir time-reversed).
__global__ void k_gru_step(int t, bf16* __restrict__ hb_base, float* __restrict__ hf_base,
                           const bf16* __restrict__ Whh_bf,
                           const float* __restrict__ bhh_f, const float* __restrict__ bhh_b,
                           const float* __restrict__ gi_all, bf16* __restrict__ hcat) {
  __shared__ bf16 As[BM * LDSP];
  __shared__ bf16 Bs[3 * BN * LDSP];
  const int dir = blockIdx.z;
  const int cur = t & 1, nxt = cur ^ 1;
  const long hoff = (long)Bsz * Hsz;
  const bf16*  hprev_b = hb_base + (dir * 2 + cur) * hoff;
  const float* hprev_f = hf_base + (dir * 2 + cur) * hoff;
  bf16*  hnext_b = hb_base + (dir * 2 + nxt) * hoff;
  float* hnext_f = hf_base + (dir * 2 + nxt) * hoff;
  const bf16* Whh = Whh_bf + (long)dir * 3 * Hsz * Hsz;
  const float* bhh = dir ? bhh_b : bhh_f;
  const float* gi = gi_all + (long)dir * MROWS * 1536 + (long)t * Bsz * 1536;
  const int m0 = blockIdx.x * BM;
  const int n0 = blockIdx.y * BN;
  const int tid  = threadIdx.x;
  const int lane = tid & 31;
  const int wave = tid >> 5;
  const int arow  = ((wave & 3) << 5) + (lane & 15);
  const int bcol  = ((wave >> 2) << 5) + (lane & 15);
  const int khalf = (lane >> 4) << 3;

  v8f acc[3][2][2];
#pragma unroll
  for (int g = 0; g < 3; ++g)
#pragma unroll
    for (int i = 0; i < 2; ++i)
#pragma unroll
      for (int j = 0; j < 2; ++j) acc[g][i][j] = vzero();

  for (int k0 = 0; k0 < Hsz; k0 += BK) {
    __syncthreads();
#if USE_ASYNC_LDS
    {  // shared A tile of h_prev (async global -> LDS, no VGPR round-trip)
      int r = tid >> 1, c = (tid & 1) << 4;
      const bf16* g = hprev_b + (long)(m0 + r) * Hsz + k0 + c;
      bf16* l = As + r * LDSP + c;
      async_copy16(g, l);
      async_copy16(g + 8, l + 8);
    }
#pragma unroll
    for (int g = 0; g < 3; ++g) {  // three Whh gate slices
      int r = tid >> 2, c = (tid & 3) << 3;
      async_copy16(Whh + (long)(g * Hsz + n0 + r) * Hsz + k0 + c,
                   Bs + (g * BN + r) * LDSP + c);
    }
    wait_async0();
#else
    {  // shared A tile of h_prev
      int r = tid >> 1, c = (tid & 1) << 4;
      const uint4* s =
          reinterpret_cast<const uint4*>(hprev_b + (long)(m0 + r) * Hsz + k0 + c);
      uint4* d = reinterpret_cast<uint4*>(As + r * LDSP + c);
      d[0] = s[0]; d[1] = s[1];
    }
#pragma unroll
    for (int g = 0; g < 3; ++g) {  // three Whh gate slices
      int r = tid >> 2, c = (tid & 3) << 3;
      *reinterpret_cast<uint4*>(Bs + (g * BN + r) * LDSP + c) =
          *reinterpret_cast<const uint4*>(Whh + (long)(g * Hsz + n0 + r) * Hsz + k0 + c);
    }
#endif
    __syncthreads();
    v16bf a0 = frag_from_lds(As + arow * LDSP, khalf);
    v16bf a1 = frag_from_lds(As + (arow + 16) * LDSP, khalf);
#pragma unroll
    for (int g = 0; g < 3; ++g) {
      v16bf b0 = frag_from_lds(Bs + (g * BN + bcol) * LDSP, khalf);
      v16bf b1 = frag_from_lds(Bs + (g * BN + bcol + 16) * LDSP, khalf);
      wmma_acc(acc[g][0][0], a0, b0);
      wmma_acc(acc[g][0][1], a0, b1);
      wmma_acc(acc[g][1][0], a1, b0);
      wmma_acc(acc[g][1][1], a1, b1);
    }
  }

#pragma unroll
  for (int i = 0; i < 2; ++i)
#pragma unroll
    for (int j = 0; j < 2; ++j) {
      int c = n0 + acc_col(tid, j);
      float bh_r = bhh[c], bh_z = bhh[Hsz + c], bh_n = bhh[2 * Hsz + c];
#pragma unroll
      for (int e = 0; e < 8; ++e) {
        int r = m0 + acc_row(tid, i, e);
        long gri = (long)r * 1536;
        float i_r = gi[gri + c];
        float i_z = gi[gri + Hsz + c];
        float i_n = gi[gri + 2 * Hsz + c];
        float rg = sigmoidf_(i_r + acc[0][i][j][e] + bh_r);
        float zg = sigmoidf_(i_z + acc[1][i][j][e] + bh_z);
        float ng = tanhf(i_n + rg * (acc[2][i][j][e] + bh_n));
        float hp = hprev_f[(long)r * Hsz + c];
        float hn = (1.f - zg) * ng + zg * hp;
        hnext_f[(long)r * Hsz + c] = hn;
        hnext_b[(long)r * Hsz + c] = (bf16)hn;
        long trow = dir ? ((long)(Tsz - 1 - t) * Bsz + r) : ((long)t * Bsz + r);
        hcat[trow * (2 * Hsz) + (long)dir * Hsz + c] = (bf16)hn;
      }
    }
}

// rnn_est = hcat @ W_proj^T + b_proj ; rnn_imp = m*x + (1-m)*rnn_est (bf16);
// accumulate per-timestep sum((est-x)^2 * m) for the RNN RMSE term.
__global__ void k_proj(const bf16* __restrict__ hcat, const bf16* __restrict__ Wp,
                       const float* __restrict__ bp, const float* __restrict__ Xf,
                       const float* __restrict__ Mf, bf16* __restrict__ rnn_imp,
                       float* __restrict__ sum_rnn) {
  __shared__ bf16 As[BM * LDSP];
  __shared__ bf16 Bs[BN * LDSP];
  long m0 = (long)blockIdx.x * BM;
  int n0 = blockIdx.y * BN;
  v8f acc[2][2];
#pragma unroll
  for (int i = 0; i < 2; ++i)
#pragma unroll
    for (int j = 0; j < 2; ++j) acc[i][j] = vzero();
  gemm_core(hcat + m0 * 1024, 1024, Wp + (long)n0 * 1024, 1024, 1024, acc, As, Bs);
  int tid = threadIdx.x;
  int t = (int)(m0 / Bsz);  // BM divides Bsz -> one timestep per block
  float lsum = 0.f;
#pragma unroll
  for (int i = 0; i < 2; ++i)
#pragma unroll
    for (int j = 0; j < 2; ++j) {
      int f = n0 + acc_col(tid, j);
      float bv = bp[f];
#pragma unroll
      for (int e = 0; e < 8; ++e) {
        long r = m0 + acc_row(tid, i, e);
        int b = (int)(r % Bsz);
        float est = acc[i][j][e] + bv;
        long xi = ((long)b * Tsz + t) * Fsz + f;
        float x = Xf[xi], m = Mf[xi];
        rnn_imp[r * Fsz + f] = (bf16)(m * x + (1.f - m) * est);
        float d = est - x;
        lsum += d * d * m;
      }
    }
  float* red = reinterpret_cast<float*>(As);
  red[tid] = lsum;
  __syncthreads();
  for (int s = 128; s > 0; s >>= 1) {
    if (tid < s) red[tid] += red[tid + s];
    __syncthreads();
  }
  if (tid == 0) atomicAdd(&sum_rnn[t], red[0]);
}

// fcn_h = tanh(x@(U*mm)^T + rnn_imp@(V1*mm)^T + m@V2^T + beta)   (bf16 out)
__global__ void k_fcn_h(const bf16* __restrict__ x_bf, const bf16* __restrict__ imp_bf,
                        const bf16* __restrict__ m_bf, const bf16* __restrict__ Um,
                        const bf16* __restrict__ V1m, const bf16* __restrict__ V2b,
                        const float* __restrict__ beta, bf16* __restrict__ fcn_h) {
  __shared__ bf16 As[BM * LDSP];
  __shared__ bf16 Bs[BN * LDSP];
  long m0 = (long)blockIdx.x * BM;
  int n0 = blockIdx.y * BN;
  v8f acc[2][2];
#pragma unroll
  for (int i = 0; i < 2; ++i)
#pragma unroll
    for (int j = 0; j < 2; ++j) acc[i][j] = vzero();
  gemm_core(x_bf + m0 * Fsz, Fsz, Um + (long)n0 * Fsz, Fsz, Fsz, acc, As, Bs);
  gemm_core(imp_bf + m0 * Fsz, Fsz, V1m + (long)n0 * Fsz, Fsz, Fsz, acc, As, Bs);
  gemm_core(m_bf + m0 * Fsz, Fsz, V2b + (long)n0 * Fsz, Fsz, Fsz, acc, As, Bs);
  int tid = threadIdx.x;
#pragma unroll
  for (int i = 0; i < 2; ++i)
#pragma unroll
    for (int j = 0; j < 2; ++j) {
      int f = n0 + acc_col(tid, j);
      float bv = beta[f];
#pragma unroll
      for (int e = 0; e < 8; ++e) {
        long r = m0 + acc_row(tid, i, e);
        fcn_h[r * Fsz + f] = (bf16)tanhf(acc[i][j][e] + bv);
      }
    }
}

// fcn_est = fcn_h @ W_fin^T + b_fin ; imputed -> d_out[B,T,F];
// accumulate per-t sum((est-x)^2*m) and global sum(|est-x|*m).
__global__ void k_fcn_est(const bf16* __restrict__ fcn_h, const bf16* __restrict__ Wfin,
                          const float* __restrict__ bfin, const float* __restrict__ Xf,
                          const float* __restrict__ Mf, float* __restrict__ imputed,
                          float* __restrict__ sum_fcn, float* __restrict__ mae_num) {
  __shared__ bf16 As[BM * LDSP];
  __shared__ bf16 Bs[BN * LDSP];
  long m0 = (long)blockIdx.x * BM;
  int n0 = blockIdx.y * BN;
  v8f acc[2][2];
#pragma unroll
  for (int i = 0; i < 2; ++i)
#pragma unroll
    for (int j = 0; j < 2; ++j) acc[i][j] = vzero();
  gemm_core(fcn_h + m0 * Fsz, Fsz, Wfin + (long)n0 * Fsz, Fsz, Fsz, acc, As, Bs);
  int tid = threadIdx.x;
  int t = (int)(m0 / Bsz);
  float lsum = 0.f, lmae = 0.f;
#pragma unroll
  for (int i = 0; i < 2; ++i)
#pragma unroll
    for (int j = 0; j < 2; ++j) {
      int f = n0 + acc_col(tid, j);
      float bv = bfin[f];
#pragma unroll
      for (int e = 0; e < 8; ++e) {
        long r = m0 + acc_row(tid, i, e);
        int b = (int)(r % Bsz);
        float est = acc[i][j][e] + bv;
        long xi = ((long)b * Tsz + t) * Fsz + f;
        float x = Xf[xi], m = Mf[xi];
        imputed[xi] = m * x + (1.f - m) * est;
        float d = est - x;
        lsum += d * d * m;
        lmae += fabsf(d) * m;
      }
    }
  float* red = reinterpret_cast<float*>(As);
  red[tid] = lsum;
  __syncthreads();
  for (int s = 128; s > 0; s >>= 1) {
    if (tid < s) red[tid] += red[tid + s];
    __syncthreads();
  }
  if (tid == 0) atomicAdd(&sum_fcn[t], red[0]);
  __syncthreads();
  red[tid] = lmae;
  __syncthreads();
  for (int s = 128; s > 0; s >>= 1) {
    if (tid < s) red[tid] += red[tid + s];
    __syncthreads();
  }
  if (tid == 0) atomicAdd(mae_num, red[0]);
}

// ------------------------- reductions --------------------------------------

__global__ void k_mask_sums(const float* __restrict__ Mf, float* __restrict__ msum,
                            float* __restrict__ mask_tot) {
  __shared__ float red[NTH];
  int bt = blockIdx.x;          // b*T + t over [B,T,F] layout
  int t = bt % Tsz;
  float v = Mf[(long)bt * Fsz + threadIdx.x];
  red[threadIdx.x] = v;
  __syncthreads();
  for (int s = 128; s > 0; s >>= 1) {
    if (threadIdx.x < s) red[threadIdx.x] += red[threadIdx.x + s];
    __syncthreads();
  }
  if (threadIdx.x == 0) {
    atomicAdd(&msum[t], red[0]);
    atomicAdd(mask_tot, red[0]);
  }
}

__global__ void k_finalize(const float* __restrict__ msum, const float* __restrict__ sum_rnn,
                           const float* __restrict__ sum_fcn, const float* __restrict__ mae_num,
                           const float* __restrict__ mask_tot, float* __restrict__ out_tail) {
  __shared__ float red[Tsz];
  int t = threadIdx.x;          // 64 threads
  float ms = msum[t] + 1e-9f;
  red[t] = sqrtf(sum_fcn[t] / ms) + sqrtf(sum_rnn[t] / ms);
  __syncthreads();
  for (int s = 32; s > 0; s >>= 1) {
    if (t < s) red[t] += red[t + s];
    __syncthreads();
  }
  if (t == 0) {
    out_tail[0] = red[0] / (float)Tsz;                 // recon_loss
    out_tail[1] = mae_num[0] / (mask_tot[0] + 1e-9f);  // recon_mae
  }
}

// ------------------------- launcher ----------------------------------------

extern "C" void kernel_launch(void* const* d_in, const int* in_sizes, int n_in,
                              void* d_out, int out_size, void* d_ws, size_t ws_size,
                              hipStream_t stream) {
  (void)in_sizes; (void)n_in; (void)out_size; (void)ws_size;
  const float* Xf     = (const float*)d_in[0];
  const float* Mf     = (const float*)d_in[1];
  const float* Df     = (const float*)d_in[2];
  const float* Xb     = (const float*)d_in[3];
  const float* Mb     = (const float*)d_in[4];
  const float* Db     = (const float*)d_in[5];
  const float* Wih_f  = (const float*)d_in[6];
  const float* Whh_f  = (const float*)d_in[7];
  const float* bih_f  = (const float*)d_in[8];
  const float* bhh_f  = (const float*)d_in[9];
  const float* Wih_b  = (const float*)d_in[10];
  const float* Whh_b  = (const float*)d_in[11];
  const float* bih_b  = (const float*)d_in[12];
  const float* bhh_b  = (const float*)d_in[13];
  const float* W_proj = (const float*)d_in[14];
  const float* b_proj = (const float*)d_in[15];
  const float* U      = (const float*)d_in[16];
  const float* V1     = (const float*)d_in[17];
  const float* V2     = (const float*)d_in[18];
  const float* beta   = (const float*)d_in[19];
  const float* W_fin  = (const float*)d_in[20];
  const float* b_fin  = (const float*)d_in[21];
  float* out = (float*)d_out;

  // workspace carve (256B aligned)
  char* p = (char*)d_ws;
  auto take = [&](size_t bytes) -> char* {
    char* q = p;
    p += (bytes + 255) & ~size_t(255);
    return q;
  };
  bf16* xs      = (bf16*)take(2L * MROWS * 768 * sizeof(bf16));
  bf16* x_bf    = (bf16*)take((long)MROWS * Fsz * sizeof(bf16));
  bf16* m_bf    = (bf16*)take((long)MROWS * Fsz * sizeof(bf16));
  bf16* rnn_imp = (bf16*)take((long)MROWS * Fsz * sizeof(bf16));
  bf16* fcn_h   = (bf16*)take((long)MROWS * Fsz * sizeof(bf16));
  float* gi     = (float*)take(2L * MROWS * 1536 * sizeof(float));
  bf16* hcat    = (bf16*)take((long)MROWS * 2 * Hsz * sizeof(bf16));
  float* hf_base = (float*)take(2L * 2 * Bsz * Hsz * sizeof(float));
  bf16*  hb_base = (bf16*)take(2L * 2 * Bsz * Hsz * sizeof(bf16));
  bf16* Wih_bf  = (bf16*)take(2L * 1536 * 768 * sizeof(bf16));
  bf16* Whh_bf  = (bf16*)take(2L * 1536 * 512 * sizeof(bf16));
  bf16* Wp_bf   = (bf16*)take(256L * 1024 * sizeof(bf16));
  bf16* Um_bf   = (bf16*)take(256L * 256 * sizeof(bf16));
  bf16* V1m_bf  = (bf16*)take(256L * 256 * sizeof(bf16));
  bf16* V2_bf   = (bf16*)take(256L * 256 * sizeof(bf16));
  bf16* Wfin_bf = (bf16*)take(256L * 256 * sizeof(bf16));
  float* sums   = (float*)take(256 * sizeof(float));
  float* msum    = sums;        // [64]
  float* sum_rnn = sums + 64;   // [64]
  float* sum_fcn = sums + 128;  // [64]
  float* mae_num  = sums + 192;
  float* mask_tot = sums + 193;

  const dim3 thr(NTH);

  // stage 0: pack inputs + convert weights + zero state/accumulators
  k_pack_xs<<<dim3(98304, 1, 2), thr, 0, stream>>>(Xf, Mf, Df, Xb, Mb, Db,
                                                   xs, x_bf, m_bf);
  k_cvt<<<4608, thr, 0, stream>>>(Wih_f, Wih_bf, 1536, 768, 0);
  k_cvt<<<4608, thr, 0, stream>>>(Wih_b, Wih_bf + 1536L * 768, 1536, 768, 0);
  k_cvt<<<3072, thr, 0, stream>>>(Whh_f, Whh_bf, 1536, 512, 0);
  k_cvt<<<3072, thr, 0, stream>>>(Whh_b, Whh_bf + 1536L * 512, 1536, 512, 0);
  k_cvt<<<1024, thr, 0, stream>>>(W_proj, Wp_bf, 256, 1024, 0);
  k_cvt<<<256,  thr, 0, stream>>>(U,  Um_bf,  256, 256, 1);
  k_cvt<<<256,  thr, 0, stream>>>(V1, V1m_bf, 256, 256, 1);
  k_cvt<<<256,  thr, 0, stream>>>(V2, V2_bf,  256, 256, 0);
  k_cvt<<<256,  thr, 0, stream>>>(W_fin, Wfin_bf, 256, 256, 0);
  k_zero<<<4096, thr, 0, stream>>>(hf_base, hb_base, sums, 256);

  // stage 1: time-parallel input GEMM (both directions)
  k_gi<<<dim3(256, 24, 2), thr, 0, stream>>>(xs, Wih_bf, bih_f, bih_b, gi);

  // stage 2: serial GRU recurrence, one fused launch per timestep
  for (int t = 0; t < Tsz; ++t)
    k_gru_step<<<dim3(4, 8, 2), thr, 0, stream>>>(t, hb_base, hf_base, Whh_bf,
                                                  bhh_f, bhh_b, gi, hcat);

  // stage 3: projection + FCN chain + losses
  k_mask_sums<<<32768, thr, 0, stream>>>(Mf, msum, mask_tot);
  k_proj<<<dim3(256, 4), thr, 0, stream>>>(hcat, Wp_bf, b_proj, Xf, Mf,
                                           rnn_imp, sum_rnn);
  k_fcn_h<<<dim3(256, 4), thr, 0, stream>>>(x_bf, rnn_imp, m_bf, Um_bf, V1m_bf,
                                            V2_bf, beta, fcn_h);
  k_fcn_est<<<dim3(256, 4), thr, 0, stream>>>(fcn_h, Wfin_bf, b_fin, Xf, Mf,
                                              out, sum_fcn, mae_num);
  k_finalize<<<1, Tsz, 0, stream>>>(msum, sum_rnn, sum_fcn, mae_num, mask_tot,
                                    out + (long)Bsz * Tsz * Fsz);
}
